// KPConvLayer_13795434955245
// MI455X (gfx1250) — matrix operand order repaired and checked by reading, not compiled, not run on hardware
//
#include <hip/hip_runtime.h>

// ---------------------------------------------------------------- constants
#define MNEI      32          // neighbors per query point
#define KP        15          // kernel points (padded to 16 in WMMA)
#define CIN       64
#define COUT      64
#define NQ_TILE   64          // query points per workgroup
#define WAVES     4           // 128 threads
#define KDIM      1024        // 16 (k-pad) * 64 (cin) flattened K for stage B
#define KSTEPS    32          // KDIM / 32
#define OTILES    4           // COUT / 16
#define INV_SIGMA 20.0f       // 1 / point_influence (0.05)

typedef __bf16        v16bf __attribute__((ext_vector_type(16)));
typedef float         v8f   __attribute__((ext_vector_type(8)));
typedef unsigned int  u32x4 __attribute__((ext_vector_type(4)));
typedef int           i32x4 __attribute__((ext_vector_type(4)));
typedef int           i32x8 __attribute__((ext_vector_type(8)));

union Frag { u32x4 u[2]; v16bf v; };

// Tensor Data Mover availability (device pass only; host pass falls back)
#if defined(__has_builtin)
#if __has_builtin(__builtin_amdgcn_tensor_load_to_lds) && \
    __has_builtin(__builtin_amdgcn_s_wait_tensorcnt)
#define HAVE_TDM 1
#endif
#endif

// ---------------------------------------------------------------- LDS layout (bytes)
// WB  : swizzled bf16 weight [OTILES][KSTEPS][32 lanes][16 elems]  = 131072
// WGT : per-point weighted rows, 64 pts x 1040 ushorts (padded)    = 133120
// STG : per-wave staging (AW 1024B + BFT 5120B)                    =  24576
#define WB_OFF        0
#define WB_BYTES      (OTILES * KSTEPS * 32 * 16 * 2)     // 131072
#define WGT_OFF       (WB_OFF + WB_BYTES)
#define WGT_STRIDE_H  1040                                 // ushorts per point row (2080B, 16B aligned, bank-skewed)
#define WGT_BYTES     (NQ_TILE * WGT_STRIDE_H * 2)         // 133120
#define STG_OFF       (WGT_OFF + WGT_BYTES)
#define AW_H          (16 * 32)                            // 512 ushorts
#define BFT_STRIDE_H  40                                   // 80B per channel row (16B aligned)
#define BFT_H         (64 * BFT_STRIDE_H)                  // 2560 ushorts
#define STG_PER_WAVE  ((AW_H + BFT_H) * 2)                 // 6144 bytes
#define LDS_TOTAL     (STG_OFF + WAVES * STG_PER_WAVE)     // 288768 bytes <= 320KB

__device__ __forceinline__ unsigned short f2bf(float f) {
  unsigned u = __float_as_uint(f);
  unsigned r = u + 0x7FFFu + ((u >> 16) & 1u);            // round-to-nearest-even
  return (unsigned short)(r >> 16);
}

// ---------------------------------------------------------------- setup: f32 weight -> bf16 swizzled for WMMA B-fragments
// element order: [o_tile t][k_step s][lane L][elem e], value = W[k][c][o]
//   kk = (L/16)*16 + e ; gk = s*32+kk ; k = gk/64 ; c = gk%64 ; o = t*16 + (L%16)
//   rows with k == 15 (pad) are zero.
__global__ __launch_bounds__(256)
void kpconv_swz_weight(const float* __restrict__ weight, unsigned short* __restrict__ wb) {
  int tid = blockIdx.x * blockDim.x + threadIdx.x;
  if (tid >= OTILES * KSTEPS * 32 * 16) return;
  int e = tid & 15;
  int L = (tid >> 4) & 31;
  int s = (tid >> 9) & (KSTEPS - 1);
  int t = tid / (512 * KSTEPS);
  int kk = ((L >> 4) << 4) + e;
  int gk = s * 32 + kk;
  int k  = gk >> 6;
  int c  = gk & 63;
  int o  = (t << 4) + (L & 15);
  float w = (k < KP) ? weight[(k * CIN + c) * COUT + o] : 0.0f;
  wb[tid] = f2bf(w);
}

// ---------------------------------------------------------------- main fused kernel
__global__ __launch_bounds__(128)
void kpconv_wmma(const float* __restrict__ qp, const float* __restrict__ sp,
                 const int* __restrict__ nbr, const float* __restrict__ x,
                 const float* __restrict__ kpts, const unsigned short* __restrict__ wbg,
                 float* __restrict__ out, int N, int N0)
{
  extern __shared__ __align__(16) char smem[];
  unsigned short* WB  = (unsigned short*)(smem + WB_OFF);
  unsigned short* WGT = (unsigned short*)(smem + WGT_OFF);

  const int tid  = threadIdx.x;
  const int wave = tid >> 5;
  const int lane = tid & 31;
  const int lrow = lane & 15;
  const int hi   = (lane >= 16) ? 1 : 0;

  unsigned short* AW  = (unsigned short*)(smem + STG_OFF + wave * STG_PER_WAVE);
  unsigned short* BFT = AW + AW_H;

  // ---- weight tile -> LDS
#ifdef HAVE_TDM
  // Tensor Data Mover: one DMA per workgroup (issued by wave 0; EXEC ignored,
  // one issue per executing wave).  D# per ISA ch.8: 1-D tile of 16384 x 8B.
  if (wave == 0) {
    unsigned long long ga = (unsigned long long)wbg;
    u32x4 g0;
    g0[0] = 1u;                                   // count=1 valid descriptor
    g0[1] = (unsigned)WB_OFF;                     // lds_addr
    g0[2] = (unsigned)(ga & 0xffffffffu);         // global_addr[31:0]
    g0[3] = (unsigned)((ga >> 32) & 0x1ffffffu)   // global_addr[56:32]
          | (2u << 30);                           // type = 2 ("image")
    i32x8 g1;
    g1[0] = 0x00030000;                           // wg_mask=0, data_size=3 (8B)
    g1[1] = (int)(16384u << 16);                  // tensor_dim0[15:0]=16384
    g1[2] = (int)(1u << 16);                      // dim0[31:16]=0, tensor_dim1=1
    g1[3] = (int)(16384u << 16);                  // dim1[31:16]=0, tile_dim0=16384
    g1[4] = 1;                                    // tile_dim1=1, tile_dim2=0
    g1[5] = 16384;                                // tensor_dim0_stride[31:0]
    g1[6] = (int)(16384u << 16);                  // dim0_stride[47:32]=0, dim1_stride[15:0]
    g1[7] = 0;
    i32x4 gz = {0, 0, 0, 0};
#if __clang_major__ >= 23
    i32x8 gz8 = {0, 0, 0, 0, 0, 0, 0, 0};
    __builtin_amdgcn_tensor_load_to_lds(g0, g1, gz, gz, gz8, 0);
#else
    __builtin_amdgcn_tensor_load_to_lds(g0, g1, gz, gz, 0);
#endif
    __builtin_amdgcn_s_wait_tensorcnt(0);
  }
#else
  {
    const u32x4* src = (const u32x4*)wbg;
    u32x4*       dst = (u32x4*)WB;
    #pragma unroll 4
    for (int i = 0; i < WB_BYTES / 16 / 128; ++i)       // 64 iters
      dst[i * 128 + tid] = src[i * 128 + tid];
  }
#endif
  __syncthreads();

  // ---- preload kernel point coordinates (uniform -> scalar loads)
  float kpx[KP], kpy[KP], kpz[KP];
  #pragma unroll
  for (int k = 0; k < KP; ++k) {
    kpx[k] = kpts[3 * k + 0];
    kpy[k] = kpts[3 * k + 1];
    kpz[k] = kpts[3 * k + 2];
  }

  // ================= Phase 1: per point, weighted[k,c] = w[k,m] @ feat[m,c] =================
  for (int pl = 0; pl < 16; ++pl) {
    int n  = blockIdx.x * NQ_TILE + wave * 16 + pl;
    int nc = (n < N) ? n : (N - 1);                     // clamped compute keeps EXEC full

    // lane = neighbor m
    int  idx   = nbr[nc * MNEI + lane];
    bool valid = (idx < N0);
    float sx = valid ? sp[idx * 3 + 0] : 1.0e6f;
    float sy = valid ? sp[idx * 3 + 1] : 1.0e6f;
    float sz = valid ? sp[idx * 3 + 2] : 1.0e6f;
    float cx = sx - qp[nc * 3 + 0];
    float cy = sy - qp[nc * 3 + 1];
    float cz = sz - qp[nc * 3 + 2];

    // influence weights -> AW[k][m] (row-major 16x32 bf16), pad row k=15 zero
    #pragma unroll
    for (int k = 0; k < 16; ++k) {
      float w = 0.0f;
      if (k < KP) {
        float dx = cx - kpx[k], dy = cy - kpy[k], dz = cz - kpz[k];
        w = fmaxf(1.0f - sqrtf(dx * dx + dy * dy + dz * dz) * INV_SIGMA, 0.0f);
      }
      AW[k * 32 + lane] = f2bf(w);
    }

    // gather features, store transposed BFT[c][m]; lane owns channels 2L,2L+1
    // (one b64 load per lane per neighbor, 256B contiguous per row -> coalesced)
    #pragma unroll 4
    for (int m = 0; m < MNEI; ++m) {
      int  si = __shfl(idx, m);
      float2 f = make_float2(0.0f, 0.0f);
      if (si < N0) f = *(const float2*)(x + (size_t)si * CIN + 2 * lane);
      BFT[(2 * lane)     * BFT_STRIDE_H + m] = f2bf(f.x);
      BFT[(2 * lane + 1) * BFT_STRIDE_H + m] = f2bf(f.y);
    }

    // A fragment: lane = row k, K(m) = 0..15 / 16..31
    Frag a;
    a.u[0] = *(const u32x4*)(AW + lrow * 32 +     (hi ? 16 : 0));
    a.u[1] = *(const u32x4*)(AW + lrow * 32 + 8 + (hi ? 16 : 0));

    unsigned short* wrow = WGT + (wave * 16 + pl) * WGT_STRIDE_H;
    #pragma unroll
    for (int t = 0; t < 4; ++t) {
      // B fragment: lane = col c, K(m) = 0..15 / 16..31 (contiguous in BFT row)
      Frag b;
      const unsigned short* bb = BFT + (t * 16 + lrow) * BFT_STRIDE_H + (hi ? 16 : 0);
      b.u[0] = *(const u32x4*)(bb);
      b.u[1] = *(const u32x4*)(bb + 8);
      v8f c = {};
      c = __builtin_amdgcn_wmma_f32_16x16x32_bf16(false, a.v, false, b.v,
                                                  (short)0, c, false, false);
      // scatter D to WGT row: element r -> (k = r + 8*hi, c = t*16 + lrow)
      #pragma unroll
      for (int r = 0; r < 8; ++r) {
        int kk = r + (hi ? 8 : 0);
        wrow[kk * 64 + t * 16 + lrow] = f2bf(c[r]);
      }
    }
  }

  // ================= Phase 2: out[16 pts, 64] = weighted[16,1024] @ Wb[1024,64] =================
  v8f acc[4] = {};
  const unsigned short* arow = WGT + (wave * 16 + lrow) * WGT_STRIDE_H;
  for (int s = 0; s < KSTEPS; ++s) {
    Frag a;  // lane = row (point), K = s*32 + 0..15 / 16..31
    a.u[0] = *(const u32x4*)(arow + s * 32 +     (hi ? 16 : 0));
    a.u[1] = *(const u32x4*)(arow + s * 32 + 8 + (hi ? 16 : 0));
    #pragma unroll
    for (int t = 0; t < 4; ++t) {
      const unsigned short* bb = WB + (t * KSTEPS + s) * 512 + lane * 16;
      Frag b;
      b.u[0] = *(const u32x4*)(bb);
      b.u[1] = *(const u32x4*)(bb + 8);
      acc[t] = __builtin_amdgcn_wmma_f32_16x16x32_bf16(false, a.v, false, b.v,
                                                       (short)0, acc[t], false, false);
    }
  }

  // ---- store: D element (t, r, lane) -> out[n = base + r + 8*hi][o = t*16 + lrow]
  #pragma unroll
  for (int t = 0; t < 4; ++t) {
    #pragma unroll
    for (int r = 0; r < 8; ++r) {
      int prow = r + (hi ? 8 : 0);
      int n = blockIdx.x * NQ_TILE + wave * 16 + prow;
      if (n < N) out[n * COUT + t * 16 + lrow] = acc[t][r];
    }
  }
}

// ---------------------------------------------------------------- host
extern "C" void kernel_launch(void* const* d_in, const int* in_sizes, int n_in,
                              void* d_out, int out_size, void* d_ws, size_t ws_size,
                              hipStream_t stream) {
  const float* qp     = (const float*)d_in[0];   // query_points  [N,3]
  const float* sp     = (const float*)d_in[1];   // support_points[N0,3]
  const int*   nbr    = (const int*)  d_in[2];   // neighbors     [N,M]
  const float* x      = (const float*)d_in[3];   // x             [N0,CIN]
  const float* kpts   = (const float*)d_in[4];   // K_points      [K,3]
  const float* weight = (const float*)d_in[5];   // weight        [K,CIN,COUT]
  float* out = (float*)d_out;

  int N  = in_sizes[2] / MNEI;
  int N0 = in_sizes[1] / 3;

  unsigned short* wb = (unsigned short*)d_ws;    // 131072 bytes of scratch
  int swz_total = OTILES * KSTEPS * 32 * 16;
  hipLaunchKernelGGL(kpconv_swz_weight, dim3((swz_total + 255) / 256), dim3(256),
                     0, stream, weight, wb);

  int blocks = (N + NQ_TILE - 1) / NQ_TILE;
  hipLaunchKernelGGL(kpconv_wmma, dim3(blocks), dim3(128), LDS_TOTAL, stream,
                     qp, sp, nbr, x, kpts, wb, out, N, N0);
}
